// SelectiveSSM_87205015978001
// MI455X (gfx1250) — matrix operand order repaired and checked by reading, not compiled, not run on hardware
//
#include <hip/hip_runtime.h>

typedef __attribute__((ext_vector_type(16))) __bf16 v16bf;
typedef __attribute__((ext_vector_type(8)))  __bf16 v8bf;
typedef __attribute__((ext_vector_type(4)))  __bf16 v4bf;
typedef __attribute__((ext_vector_type(8)))  float  v8f;
typedef __attribute__((ext_vector_type(4)))  float  f32x4;

#define BATCH 8
#define SEQ   4096
#define DM    1024
#define NS    16
#define MROWS (BATCH * SEQ)          // 32768 rows of x
#define KTILES (DM / 32)             // 32 k-steps of K=32

// ---------------------------------------------------------------------------
// Pack x (f32 row-major 32768x1024) -> bf16 row-major (one pass, amortized
// over 66 column-tiles of reuse in the projections).
// ---------------------------------------------------------------------------
__global__ __launch_bounds__(256) void pack_x_kernel(
    const float* __restrict__ x, __bf16* __restrict__ xh) {
  size_t e = ((size_t)blockIdx.x * 256 + threadIdx.x) * 4;
  if (e >= (size_t)MROWS * DM) return;
  f32x4 v = *(const f32x4*)(x + e);
  v4bf o;
#pragma unroll
  for (int i = 0; i < 4; ++i) o[i] = (__bf16)v[i];
  *(v4bf*)(xh + e) = o;
}

// Pack Wdelta (f32, (in=K 1024) x (out=N 1024)) into the WMMA B-fragment
// layout: wp[kt][n][kk], kk = 0..31 contiguous (lane n, half h loads 16
// contiguous bf16 at n*32 + h*16).
__global__ __launch_bounds__(256) void pack_wdelta_kernel(
    const float* __restrict__ W, __bf16* __restrict__ wp) {
  int idx = blockIdx.x * 256 + threadIdx.x;            // 1M elements
  if (idx >= DM * DM) return;
  int kk = idx & 31;
  int n  = (idx >> 5) & (DM - 1);
  int kt = idx >> 15;
  wp[idx] = (__bf16)W[(size_t)(kt * 32 + kk) * DM + n];
}

// Pack [Wb | Wc] (each K=1024 x N=16) into bcp[kt][n(0..31)][kk(0..31)]
__global__ __launch_bounds__(256) void pack_wbc_kernel(
    const float* __restrict__ Wb, const float* __restrict__ Wc,
    __bf16* __restrict__ bcp) {
  int idx = blockIdx.x * 256 + threadIdx.x;            // 32768 elements
  if (idx >= KTILES * 32 * 32) return;
  int kk = idx & 31;
  int n  = (idx >> 5) & 31;
  int kt = idx >> 10;
  int k  = kt * 32 + kk;
  float v = (n < 16) ? Wb[(size_t)k * NS + n] : Wc[(size_t)k * NS + (n - 16)];
  bcp[idx] = (__bf16)v;
}

// ---------------------------------------------------------------------------
// Fragment builders (ISA 7.12.2 lane/row layouts, but k contiguous (h*16+s) in A and B; wave32)
// A (16x32 bf16): lane row m = lane&15, half h = lane>>4;
//   elems 0..7 = K kb+h*16.., elems 8..15 = K kb+h*16+8..  (two 16B loads)
// B (32x16 bf16): lane col n = lane&15; elems 0..15 = K kb+h*16..+15 (packed)
// C/D (16x16 f32): lane col n = lane&15; acc[i] -> row m = (lane>>4)*8 + i
// ---------------------------------------------------------------------------
__device__ __forceinline__ v16bf load_a_frag(const __bf16* __restrict__ arow,
                                             int kb, int h) {
  const v8bf lo = *(const v8bf*)(arow + kb + h * 16);
  const v8bf hi = *(const v8bf*)(arow + kb + h * 16 + 8);
  v16bf a;
#pragma unroll
  for (int i = 0; i < 8; ++i) { a[i] = lo[i]; a[8 + i] = hi[i]; }
  return a;
}

__device__ __forceinline__ v16bf load_b_frag(const __bf16* __restrict__ bp) {
  const v8bf b0 = *(const v8bf*)(bp);
  const v8bf b1 = *(const v8bf*)(bp + 8);
  v16bf b;
#pragma unroll
  for (int i = 0; i < 8; ++i) { b[i] = b0[i]; b[8 + i] = b1[i]; }
  return b;
}

__device__ __forceinline__ float softplus_f(float z) {
  return fmaxf(z, 0.0f) + log1pf(__expf(-fabsf(z)));
}

// ---------------------------------------------------------------------------
// delta = softplus(x @ Wdelta + bdelta).  M=32768, N=1024, K=1024.
// One wave per 16x64 strip (4 accumulators share the A fragment).
// 2048 m-tiles * 16 n-groups = 32768 waves, 8 waves / block.
// ---------------------------------------------------------------------------
__global__ __launch_bounds__(256) void gemm_delta_kernel(
    const __bf16* __restrict__ xh, const __bf16* __restrict__ wp,
    const float* __restrict__ bdelta, float* __restrict__ delta) {
  const int wave = blockIdx.x * 8 + (threadIdx.x >> 5);
  const int lane = threadIdx.x & 31;
  const int mt = wave >> 4;          // 0..2047
  const int ng = wave & 15;          // 0..15 -> n-tiles 4*ng .. 4*ng+3
  const int ln = lane & 15;
  const int h  = lane >> 4;

  const __bf16* arow  = xh + (size_t)(mt * 16 + ln) * DM;
  const __bf16* bbase = wp + (size_t)(ng * 64 + ln) * 32 + h * 16;

  v8f acc[4] = {{}, {}, {}, {}};
#pragma unroll 2
  for (int kt = 0; kt < KTILES; ++kt) {
    const v16bf a = load_a_frag(arow, kt * 32, h);
    const __bf16* bp = bbase + (size_t)kt * (DM * 32);
#pragma unroll
    for (int t = 0; t < 4; ++t) {
      const v16bf b = load_b_frag(bp + t * (16 * 32));
      acc[t] = __builtin_amdgcn_wmma_f32_16x16x32_bf16(
          false, a, false, b, (short)0, acc[t], false, false);
    }
  }

#pragma unroll
  for (int t = 0; t < 4; ++t) {
    const int n = ng * 64 + t * 16 + ln;
    const float bias = bdelta[n];
#pragma unroll
    for (int i = 0; i < 8; ++i) {
      const int m = mt * 16 + h * 8 + i;
      delta[(size_t)m * DM + n] = softplus_f(acc[t][i] + bias);
    }
  }
}

// ---------------------------------------------------------------------------
// BC = x @ [Wb|Wc] + [bb|bc].  M=32768, N=32, K=1024 -> bcbuf[m][0..31]
// One wave per 16x32 strip (2 accumulators).  2048 waves, 8 / block.
// ---------------------------------------------------------------------------
__global__ __launch_bounds__(256) void gemm_bc_kernel(
    const __bf16* __restrict__ xh, const __bf16* __restrict__ bcp,
    const float* __restrict__ bb, const float* __restrict__ bc,
    float* __restrict__ bcbuf) {
  const int wave = blockIdx.x * 8 + (threadIdx.x >> 5);
  const int lane = threadIdx.x & 31;
  const int mt = wave;               // 0..2047
  const int ln = lane & 15;
  const int h  = lane >> 4;

  const __bf16* arow  = xh + (size_t)(mt * 16 + ln) * DM;
  const __bf16* bbase = bcp + (size_t)ln * 32 + h * 16;

  v8f acc[2] = {{}, {}};
#pragma unroll 2
  for (int kt = 0; kt < KTILES; ++kt) {
    const v16bf a = load_a_frag(arow, kt * 32, h);
    const __bf16* bp = bbase + (size_t)kt * (32 * 32);
#pragma unroll
    for (int t = 0; t < 2; ++t) {
      const v16bf b = load_b_frag(bp + t * (16 * 32));
      acc[t] = __builtin_amdgcn_wmma_f32_16x16x32_bf16(
          false, a, false, b, (short)0, acc[t], false, false);
    }
  }

#pragma unroll
  for (int t = 0; t < 2; ++t) {
    const int n = t * 16 + ln;                      // 0..31
    const float bias = (t == 0) ? bb[ln] : bc[ln];
#pragma unroll
    for (int i = 0; i < 8; ++i) {
      const int m = mt * 16 + h * 8 + i;
      bcbuf[(size_t)m * 32 + n] = acc[t][i] + bias;
    }
  }
}

// ---------------------------------------------------------------------------
// Sequential selective scan.  One thread per (batch, channel d); h[16] in
// registers; 8-timestep chunks staged into LDS with CDNA5 async-to-LDS
// b128 loads (ASYNCcnt) — no VGPR round trip for staging.
// ---------------------------------------------------------------------------
#define TCHUNK 8
__global__ __launch_bounds__(256) void ssm_scan_kernel(
    const float* __restrict__ x, const float* __restrict__ delta,
    const float* __restrict__ bcbuf, const float* __restrict__ A_log,
    const float* __restrict__ D_param, float* __restrict__ y) {
  __shared__ __align__(16) float sD[TCHUNK * 256];
  __shared__ __align__(16) float sX[TCHUNK * 256];
  __shared__ __align__(16) float sBC[TCHUNK * 32];

  const int tid   = threadIdx.x;
  const int b     = blockIdx.x >> 2;             // 0..7
  const int dbase = (blockIdx.x & 3) << 8;       // 0,256,512,768
  const int d     = dbase + tid;

  float h[NS];
  float Ar[NS];
#pragma unroll
  for (int n = 0; n < NS; ++n) {
    h[n]  = 0.0f;
    Ar[n] = -__expf(A_log[(size_t)d * NS + n]);
  }
  const float Dp = D_param[d];
  const size_t rowBase = (size_t)b * SEQ;

  // Per-lane async-load operands (constant across chunks).
  // 8 rows x 256 floats staged as 2 b128 issues/thread: thread covers
  // row j = tid>>6 (+4 for second issue), 4 consecutive floats at (tid&63)*4.
  const int jq = tid >> 6;
  const int lq = tid & 63;
  const unsigned voffDX0 = (unsigned)((jq * DM + dbase + lq * 4) * 4);
  const unsigned voffDX1 = voffDX0 + 4u * DM * 4u;
  const unsigned ldsD0 = (unsigned)(unsigned long long)&sD[jq * 256 + lq * 4];
  const unsigned ldsD1 = ldsD0 + 4u * 256u * 4u;
  const unsigned ldsX0 = (unsigned)(unsigned long long)&sX[jq * 256 + lq * 4];
  const unsigned ldsX1 = ldsX0 + 4u * 256u * 4u;
  // BC: 8 rows x 32 floats = 64 b128 issues (threads 0..63 only)
  const unsigned voffBC = (unsigned)((((tid >> 3) * 32) + (tid & 7) * 4) * 4);
  const unsigned ldsBC  = (unsigned)(unsigned long long)&sBC[0] + (unsigned)(tid * 16);

  for (int c = 0; c < SEQ / TCHUNK; ++c) {
    const int t0 = c * TCHUNK;
    const float* sadD  = delta + (rowBase + t0) * DM;
    const float* sadX  = x + (rowBase + t0) * DM;
    const float* sadBC = bcbuf + (rowBase + t0) * 32;

    asm volatile("global_load_async_to_lds_b128 %0, %1, %2"
                 :: "v"(ldsD0), "v"(voffDX0), "s"(sadD) : "memory");
    asm volatile("global_load_async_to_lds_b128 %0, %1, %2"
                 :: "v"(ldsD1), "v"(voffDX1), "s"(sadD) : "memory");
    asm volatile("global_load_async_to_lds_b128 %0, %1, %2"
                 :: "v"(ldsX0), "v"(voffDX0), "s"(sadX) : "memory");
    asm volatile("global_load_async_to_lds_b128 %0, %1, %2"
                 :: "v"(ldsX1), "v"(voffDX1), "s"(sadX) : "memory");
    if (tid < 64) {
      asm volatile("global_load_async_to_lds_b128 %0, %1, %2"
                   :: "v"(ldsBC), "v"(voffBC), "s"(sadBC) : "memory");
    }
    asm volatile("s_wait_asynccnt 0" ::: "memory");
    __syncthreads();

#pragma unroll
    for (int j = 0; j < TCHUNK; ++j) {
      const float dl = sD[j * 256 + tid];
      const float xv = sX[j * 256 + tid];
      const float dx = dl * xv;
      const float* bcrow = sBC + j * 32;
      float acc = Dp * xv;
#pragma unroll
      for (int n = 0; n < NS; ++n) {
        const float ad = __expf(dl * Ar[n]);          // exp(delta * A)
        h[n] = fmaf(ad, h[n], dx * bcrow[n]);         // A_disc*h + dt*B*x
        acc  = fmaf(h[n], bcrow[16 + n], acc);        // + h . C
      }
      y[(rowBase + t0 + j) * DM + d] = acc;
    }
    __syncthreads();
  }
}

// ---------------------------------------------------------------------------
// Host-side launch.  Inputs: x, Wb, bb, Wc, bc, Wdelta, bdelta, A_log, D_param
// ---------------------------------------------------------------------------
extern "C" void kernel_launch(void* const* d_in, const int* in_sizes, int n_in,
                              void* d_out, int out_size, void* d_ws, size_t ws_size,
                              hipStream_t stream) {
  (void)in_sizes; (void)n_in; (void)out_size; (void)ws_size;
  const float* x      = (const float*)d_in[0];
  const float* Wb     = (const float*)d_in[1];
  const float* bb     = (const float*)d_in[2];
  const float* Wc     = (const float*)d_in[3];
  const float* bc     = (const float*)d_in[4];
  const float* Wdelta = (const float*)d_in[5];
  const float* bdelta = (const float*)d_in[6];
  const float* A_log  = (const float*)d_in[7];
  const float* D_par  = (const float*)d_in[8];
  float* y = (float*)d_out;

  // workspace layout (bytes)
  char* base = (char*)d_ws;
  __bf16* wpack  = (__bf16*)(base);                          //  2 MB
  __bf16* bcpack = (__bf16*)(base + 2097152);                // 64 KB
  __bf16* xh     = (__bf16*)(base + 2162688);                // 64 MB
  float*  deltab = (float*)(base + 69271552);                // 128 MB
  float*  bcbuf  = (float*)(base + 203489280);               //   4 MB

  pack_x_kernel<<<((size_t)MROWS * DM / 4) / 256, 256, 0, stream>>>(x, xh);
  pack_wdelta_kernel<<<(DM * DM) / 256, 256, 0, stream>>>(Wdelta, wpack);
  pack_wbc_kernel<<<(KTILES * 32 * 32) / 256, 256, 0, stream>>>(Wb, Wc, bcpack);

  // delta GEMM: 2048 m-tiles * 16 n-groups waves / 8 waves per block
  gemm_delta_kernel<<<(2048 * 16) / 8, 256, 0, stream>>>(xh, wpack, bdelta, deltab);
  // B/C GEMM: 2048 waves / 8 per block
  gemm_bc_kernel<<<2048 / 8, 256, 0, stream>>>(xh, bcpack, bb, bc, bcbuf);

  // scan: 8 batches * 4 channel-chunks of 256
  ssm_scan_kernel<<<32, 256, 0, stream>>>(x, deltab, bcbuf, A_log, D_par, y);
}
